// MSA_27814208209356
// MI455X (gfx1250) — matrix operand order repaired and checked by reading, not compiled, not run on hardware
//
#include <hip/hip_runtime.h>
#include <hip/hip_bf16.h>

// ---------------------------------------------------------------------------
// Types for gfx1250 WMMA (wave32): v16h A/B fragments, v8f C/D fragment.
// ---------------------------------------------------------------------------
typedef _Float16 half_t;
typedef __attribute__((ext_vector_type(16))) _Float16 v16h;
typedef __attribute__((ext_vector_type(8)))  float    v8f;
typedef __attribute__((ext_vector_type(4)))  unsigned u32x4;
typedef int v4i_async __attribute__((vector_size(16)));   // matches builtin param

#define B_SZ   2
#define NSEQ   2048
#define DMODEL 1024
#define NHEAD  16
#define DH     64
#define N3D    3072   // 3*DMODEL
#define NCHUNK (NSEQ / 64)

// CDNA5 async memory->LDS path (ASYNCcnt). Guarded so compilation can't break
// if this toolchain doesn't expose the builtins; fallback is the same
// double-buffered pipeline with synchronous b128 stores.
#if defined(__gfx1250__) && \
    __has_builtin(__builtin_amdgcn_global_load_async_to_lds_b128) && \
    __has_builtin(__builtin_amdgcn_s_wait_asynccnt)
#define USE_ASYNC_LDS 1
#else
#define USE_ASYNC_LDS 0
#endif

// Builtin signature (from hipcc diagnostic):
//   param0: v4i addrspace(1)* (global source, printed as "__device__")
//   param1: v4i addrspace(3)* (LDS destination)
//   param2/3: imm int offset, imm int cpol
#define ASYNC_SRC(p) ((__attribute__((address_space(1))) v4i_async*)(p))
#define ASYNC_DST(p) ((__attribute__((address_space(3))) v4i_async*)(p))

static __device__ inline v8f v8f_zero() {
  v8f z = {0.f,0.f,0.f,0.f,0.f,0.f,0.f,0.f};
  return z;
}

static __device__ inline v8f wmma_f16(v16h a, v16h b, v8f c) {
  // D = A(16x32 f16) * B(32x16 f16) + C(16x16 f32)
  return __builtin_amdgcn_wmma_f32_16x16x32_f16(
      /*neg_a=*/false, a, /*neg_b=*/false, b,
      /*c_mod=*/(short)0, c, /*reuse_a=*/false, /*reuse_b=*/false);
}

// A-fragment (16x32, MxK), contiguous: lane<16 -> K = d0+{0..15},
// lane>=16 -> K = d0+{16..31}. rowbase = row M of this lane.
static __device__ inline v16h load_frag_a(const half_t* rowbase, int d0, int lane) {
  const int off = (lane < 16) ? 0 : 16;
  union { v16h v; u32x4 q[2]; } u;
  u.q[0] = *(const u32x4*)(rowbase + d0 + off);
  u.q[1] = *(const u32x4*)(rowbase + d0 + off + 8);
  return u.v;
}

// B-fragment (32x16, KxN), contiguous K order like load_frag_a: lane = column N, lane half selects
// K 0..15 vs 16..31. rowbase points at the contiguous K-run for this lane's
// column (row-major weight / K tile).
static __device__ inline v16h load_frag_b_contig(const half_t* rowbase, int k0, int lane) {
  const int off = k0 + ((lane < 16) ? 0 : 16);
  union { v16h v; u32x4 q[2]; } u;
  u.q[0] = *(const u32x4*)(rowbase + off);
  u.q[1] = *(const u32x4*)(rowbase + off + 8);
  return u.v;
}

// ---------------------------------------------------------------------------
// Kernel 1: fp32 -> f16 pack of z and W.
// ---------------------------------------------------------------------------
__global__ void pack_f16_kernel(const float* __restrict__ z, const float* __restrict__ W,
                                half_t* __restrict__ zb, half_t* __restrict__ Wb) {
  const int stride = gridDim.x * blockDim.x;
  const int nz = B_SZ * NSEQ * DMODEL;
  const int nw = N3D * DMODEL;
  for (int i = blockIdx.x * blockDim.x + threadIdx.x; i < nz; i += stride)
    zb[i] = (half_t)z[i];
  for (int i = blockIdx.x * blockDim.x + threadIdx.x; i < nw; i += stride)
    Wb[i] = (half_t)W[i];
}

// ---------------------------------------------------------------------------
// Kernel 2: qkv = z @ W^T, output in head-major f16 layout
//           qkv[b][h][m(0=q,1=k,2=v)][n][64]
// Block = 128 threads (4 waves). Wave computes a 16(M) x 64(N) strip.
// grid.x = 4096/16 = 256 (M tiles), grid.y = 3072/256 = 12 (N super-tiles).
// ---------------------------------------------------------------------------
__global__ void qkv_gemm_kernel(const half_t* __restrict__ zb,
                                const half_t* __restrict__ Wb,
                                half_t* __restrict__ qkv) {
  const int lane = threadIdx.x & 31;
  const int wv   = threadIdx.x >> 5;                 // 0..3
  const int m0   = blockIdx.x * 16;                  // row tile (b*n flat)
  const int n0   = blockIdx.y * 256 + wv * 64;       // col strip base

  const half_t* arow = zb + (size_t)(m0 + (lane & 15)) * DMODEL;

  v8f acc[4];
#pragma unroll
  for (int t = 0; t < 4; ++t) acc[t] = v8f_zero();

  for (int k0 = 0; k0 < DMODEL; k0 += 32) {
    const v16h a = load_frag_a(arow, k0, lane);
#pragma unroll
    for (int t = 0; t < 4; ++t) {
      const half_t* brow = Wb + (size_t)(n0 + t * 16 + (lane & 15)) * DMODEL;
      const v16h bf = load_frag_b_contig(brow, k0, lane);
      acc[t] = wmma_f16(a, bf, acc[t]);
    }
  }

  // Store into head-major qkv. C layout: VGPR r holds row (r or r+8 by half),
  // lane&15 is the column.
  const int halfsel = (lane >= 16) ? 8 : 0;
  const int bidx  = m0 >> 11;            // 2048 rows per batch; tile never splits
  const int nrow0 = m0 & (NSEQ - 1);
#pragma unroll
  for (int t = 0; t < 4; ++t) {
    const int n_abs = n0 + t * 16 + (lane & 15);
    const int h  = n_abs / 192;
    const int rm = n_abs % 192;
    const int mq = rm / 64;
    const int d  = rm % 64;
    half_t* dst = qkv + (((size_t)(bidx * NHEAD + h) * 3 + mq) * NSEQ) * DH + d;
#pragma unroll
    for (int r = 0; r < 8; ++r) {
      const int nrow = nrow0 + r + halfsel;
      dst[(size_t)nrow * DH] = (half_t)acc[t][r];
    }
  }
}

// ---------------------------------------------------------------------------
// Kernel 3: flash attention per (b, h). Block = 128 threads (4 waves), each
// wave owns 16 query rows; workgroup owns a 64-row Q tile. K/V streamed in
// 64-key chunks through double-buffered LDS using CDNA5 async LDS loads
// (ASYNCcnt) so the chunk j+1 copy overlaps chunk j's WMMAs.
// grid = (32 q-tiles, 16 heads, 2 batches)
// ---------------------------------------------------------------------------
__global__ void attn_kernel(const half_t* __restrict__ qkv, float* __restrict__ out) {
  const int lane = threadIdx.x & 31;
  const int wv   = threadIdx.x >> 5;   // 0..3
  const int qt   = blockIdx.x;         // 0..31
  const int h    = blockIdx.y;         // 0..15
  const int b    = blockIdx.z;         // 0..1

  const size_t headbase = ((size_t)(b * NHEAD + h) * 3) * NSEQ * DH;
  const half_t* Qp = qkv + headbase;                    // [2048][64]
  const half_t* Kp = qkv + headbase + (size_t)NSEQ * DH;
  const half_t* Vp = qkv + headbase + (size_t)2 * NSEQ * DH;

  __shared__ __align__(16) half_t Kl[2][64][72];    // padded stride: 144B
  __shared__ __align__(16) half_t Vl[2][64][72];
  __shared__ __align__(16) half_t Pl[4][16][72];    // per-wave P strip

  const int mlane   = lane & 15;
  const int halfsel = (lane >= 16) ? 8 : 0;
  const int q0      = qt * 64 + wv * 16;

  // Issue one 64-key K/V chunk copy into LDS buffer `buf`.
  auto issue_chunk = [&](int j0, int buf) {
    const u32x4* ks = (const u32x4*)(Kp + (size_t)j0 * DH);
    const u32x4* vs = (const u32x4*)(Vp + (size_t)j0 * DH);
#pragma unroll
    for (int t = 0; t < 4; ++t) {
      const int idx = (int)threadIdx.x + t * 128;   // 0..511 u32x4's
      const int row = idx >> 3;
      const int c   = idx & 7;
#if USE_ASYNC_LDS
      __builtin_amdgcn_global_load_async_to_lds_b128(
          ASYNC_SRC(ks + idx), ASYNC_DST(&Kl[buf][row][c * 8]), 0, 0);
      __builtin_amdgcn_global_load_async_to_lds_b128(
          ASYNC_SRC(vs + idx), ASYNC_DST(&Vl[buf][row][c * 8]), 0, 0);
#else
      *(u32x4*)&Kl[buf][row][c * 8] = ks[idx];
      *(u32x4*)&Vl[buf][row][c * 8] = vs[idx];
#endif
    }
  };

  // Q strip (16 rows x 64) as two A fragments, resident for whole kernel.
  const half_t* qrow = Qp + (size_t)(q0 + mlane) * DH;
  const v16h qa0 = load_frag_a(qrow, 0, lane);
  const v16h qa1 = load_frag_a(qrow, 32, lane);

  v8f O[4];
  float mrow[8], lrow[8];
#pragma unroll
  for (int t = 0; t < 4; ++t) O[t] = v8f_zero();
#pragma unroll
  for (int r = 0; r < 8; ++r) { mrow[r] = -3.0e38f; lrow[r] = 0.0f; }

  issue_chunk(0, 0);   // prologue prefetch

  for (int chunk = 0; chunk < NCHUNK; ++chunk) {
    const int buf = chunk & 1;

#if USE_ASYNC_LDS
    __builtin_amdgcn_s_wait_asynccnt(0);   // this wave's copies for `buf` done
#endif
    __syncthreads();                        // all waves' copies done; all waves
                                            // finished reading buf^1 last iter
    if (chunk + 1 < NCHUNK)
      issue_chunk((chunk + 1) * 64, buf ^ 1);   // overlaps compute below

    // S strip (16 x 64) = Q (16x64) . K_chunk^T : WMMA K-dim = head dim.
    v8f S[4];
#pragma unroll
    for (int t = 0; t < 4; ++t) {
      S[t] = v8f_zero();
      const half_t* krow = &Kl[buf][t * 16 + mlane][0];   // key = B column
      const v16h b0 = load_frag_b_contig(krow, 0, lane);
      const v16h b1 = load_frag_b_contig(krow, 32, lane);
      S[t] = wmma_f16(qa0, b0, S[t]);
      S[t] = wmma_f16(qa1, b1, S[t]);
    }

    // scale 1/sqrt(64)
#pragma unroll
    for (int t = 0; t < 4; ++t)
#pragma unroll
      for (int r = 0; r < 8; ++r) S[t][r] = S[t][r] * 0.125f;

    // Row max of chunk (across 4 tiles then across the 16-lane half).
    float mch[8];
#pragma unroll
    for (int r = 0; r < 8; ++r) {
      float v = S[0][r];
      v = fmaxf(v, S[1][r]); v = fmaxf(v, S[2][r]); v = fmaxf(v, S[3][r]);
#pragma unroll
      for (int msk = 1; msk <= 8; msk <<= 1)
        v = fmaxf(v, __shfl_xor(v, msk, 32));
      mch[r] = v;
    }

    // Online softmax update.
    float alpha[8];
#pragma unroll
    for (int r = 0; r < 8; ++r) {
      const float mnew = fmaxf(mrow[r], mch[r]);
      alpha[r] = __expf(mrow[r] - mnew);
      mrow[r] = mnew;
    }
    float sch[8];
#pragma unroll
    for (int r = 0; r < 8; ++r) sch[r] = 0.0f;
#pragma unroll
    for (int t = 0; t < 4; ++t)
#pragma unroll
      for (int r = 0; r < 8; ++r) {
        const float p = __expf(S[t][r] - mrow[r]);
        S[t][r] = p;
        sch[r] += p;
      }
#pragma unroll
    for (int r = 0; r < 8; ++r) {
#pragma unroll
      for (int msk = 1; msk <= 8; msk <<= 1)
        sch[r] += __shfl_xor(sch[r], msk, 32);
      lrow[r] = lrow[r] * alpha[r] + sch[r];
    }
#pragma unroll
    for (int t = 0; t < 4; ++t)
#pragma unroll
      for (int r = 0; r < 8; ++r) O[t][r] = O[t][r] * alpha[r];

    // C-layout -> A-layout for P via wave-private LDS strip.
#pragma unroll
    for (int t = 0; t < 4; ++t)
#pragma unroll
      for (int r = 0; r < 8; ++r)
        Pl[wv][r + halfsel][t * 16 + mlane] = (half_t)S[t][r];
    // (same-wave DS ordering; backend inserts s_wait_dscnt)

    const half_t* prow = &Pl[wv][mlane][0];
    const v16h pa0 = load_frag_a(prow, 0, lane);
    const v16h pa1 = load_frag_a(prow, 32, lane);

    // O += P (16x64) . V_chunk (64x64): B column = head-dim d, K = key.
#pragma unroll
    for (int t = 0; t < 4; ++t) {
      const int d = t * 16 + mlane;
      v16h vb0, vb1;
      {
        const int k0 = (lane < 16) ? 0 : 16;
#pragma unroll
        for (int e = 0; e < 16; ++e) vb0[e] = Vl[buf][k0 + e][d];
        const int k1 = 32 + ((lane < 16) ? 0 : 16);
#pragma unroll
        for (int e = 0; e < 16; ++e) vb1[e] = Vl[buf][k1 + e][d];
      }
      O[t] = wmma_f16(pa0, vb0, O[t]);
      O[t] = wmma_f16(pa1, vb1, O[t]);
    }
  }

  // Epilogue: O / l, store fp32 out[b][n][h*64+d].
  float rinv[8];
#pragma unroll
  for (int r = 0; r < 8; ++r) rinv[r] = 1.0f / lrow[r];
#pragma unroll
  for (int t = 0; t < 4; ++t) {
    const int d = t * 16 + mlane;
#pragma unroll
    for (int r = 0; r < 8; ++r) {
      const int row = q0 + r + halfsel;
      out[((size_t)(b * NSEQ + row)) * DMODEL + h * DH + d] = O[t][r] * rinv[r];
    }
  }
}

// ---------------------------------------------------------------------------
// Launch
// ---------------------------------------------------------------------------
extern "C" void kernel_launch(void* const* d_in, const int* in_sizes, int n_in,
                              void* d_out, int out_size, void* d_ws, size_t ws_size,
                              hipStream_t stream) {
  const float* z = (const float*)d_in[0];   // [2,2048,1024]
  const float* W = (const float*)d_in[1];   // [3072,1024]
  float* out = (float*)d_out;               // [2,2048,1024]

  half_t* zb  = (half_t*)d_ws;                              // 4096*1024 f16
  half_t* Wb  = zb + (size_t)B_SZ * NSEQ * DMODEL;          // 3072*1024 f16
  half_t* qkv = Wb + (size_t)N3D * DMODEL;                  // 2*16*3*2048*64 f16

  pack_f16_kernel<<<2048, 256, 0, stream>>>(z, W, zb, Wb);
  qkv_gemm_kernel<<<dim3(256, 12), 128, 0, stream>>>(zb, Wb, qkv);
  attn_kernel<<<dim3(32, 16, 2), 128, 0, stream>>>(qkv, out);
}
